// HomogeneousGatingNetwork_23957327577727
// MI455X (gfx1250) — matrix operand order repaired and checked by reading, not compiled, not run on hardware
//
#include <hip/hip_runtime.h>
#include <math.h>

// Problem constants (match the reference)
#define NB 16384   // batch
#define SL 60      // sequence length
#define NV 40      // vocab
#define FD 168     // feature dim
#define NE 4       // experts

typedef __attribute__((ext_vector_type(16))) __bf16          v16bf;
typedef __attribute__((ext_vector_type(8)))  float           v8f;
typedef __attribute__((ext_vector_type(16))) unsigned short  v16us;
typedef __attribute__((ext_vector_type(8)))  unsigned short  v8us;

static __device__ __forceinline__ unsigned short f2bf(float f) {
  union { float f; unsigned u; } v; v.f = f;
  unsigned u = v.u + 0x7fffu + ((v.u >> 16) & 1u);   // round-to-nearest-even
  return (unsigned short)(u >> 16);
}
static __device__ __forceinline__ float bf2f(unsigned short h) {
  union { unsigned u; float f; } v; v.u = ((unsigned)h) << 16;
  return v.f;
}
static __device__ __forceinline__ v16bf as_bf16(v16us u) {
  union { v16us u; v16bf b; } v; v.u = u; return v.b;
}
static __device__ __forceinline__ float sigmoid_(float x) {
  return 1.0f / (1.0f + __expf(-x));
}
static __device__ __forceinline__ float tanh_(float x) {
  x = fminf(fmaxf(x, -15.f), 15.f);
  float t = __expf(2.0f * x);
  return (t - 1.0f) / (t + 1.0f);
}

// ---------------------------------------------------------------------------
// Kernel 0: fold embedding + input projection + both biases into a 40x256 LUT
//   T[v][j] = sum_k emb[v][k]*wih[k][j] + bih[j] + bhh[j]
// ---------------------------------------------------------------------------
__global__ void build_table_kernel(const float* __restrict__ emb,
                                   const float* __restrict__ wih,
                                   const float* __restrict__ bih,
                                   const float* __restrict__ bhh,
                                   float* __restrict__ T) {
  int v = blockIdx.x;       // 40 blocks
  int j = threadIdx.x;      // 256 threads
  float s = bih[j] + bhh[j];
  for (int k = 0; k < 128; ++k) s += emb[v * 128 + k] * wih[k * 256 + j];
  T[v * 256 + j] = s;
}

// ---------------------------------------------------------------------------
// Kernel 1: fused biLSTM. grid = (NB/128, 2 dirs), block = 256 (8 waves).
// Each wave owns 16 batch rows; c lives in registers; h lives in per-wave
// LDS as bf16 (the WMMA operand precision anyway). Per step, each wave does
// 16 N-tiles x 2 K-chunks of v_wmma_f32_16x16x32_bf16 with the token-table
// value as the C operand (D = h*whh + T[tok]).
// ---------------------------------------------------------------------------
#define LSTM_LDS (32768 + 40960 + 30720 + 16384)

__global__ __launch_bounds__(256) void lstm_kernel(
    const int* __restrict__ x, const float* __restrict__ whhF,
    const float* __restrict__ whhB, const float* __restrict__ Tf,
    const float* __restrict__ Tb, float* __restrict__ feats) {
  extern __shared__ char smem[];
  unsigned short* wfrag = (unsigned short*)smem;                 // [2][16][32][16] bf16 B-frags
  float* Tl = (float*)(smem + 32768);                            // [40][256] token table
  int* xt = (int*)(smem + 32768 + 40960);                        // [128][60] tokens
  unsigned short* hbuf = (unsigned short*)(smem + 32768 + 40960 + 30720); // [128][64] h (bf16)

  const int dir = blockIdx.y;
  const float* whh = dir ? whhB : whhF;
  const float* T   = dir ? Tb   : Tf;
  const int tid = threadIdx.x;
  const int wave = tid >> 5, lane = tid & 31, half = lane >> 4, ln = lane & 15;
  const int rowbase = blockIdx.x * 128;

  // Stage whh as pre-swizzled bf16 B-fragments: element e <-> K = half*8+e (e<8), 16+half*8+(e-8) (e>=8),
  // lane&15 <-> column. One contiguous 32B load per lane per fragment.
  for (int i = tid; i < 2 * 16 * 32 * 16; i += 256) {
    int e = i & 15, l = (i >> 4) & 31, nt = (i >> 9) & 15, kc = i >> 13;
    int k = kc * 32 + ((e < 8) ? ((l >> 4) * 8 + e) : (16 + (l >> 4) * 8 + (e - 8)));
    int n = nt * 16 + (l & 15);
    wfrag[i] = f2bf(whh[k * 256 + n]);
  }
  for (int i = tid; i < NV * 256; i += 256) Tl[i] = T[i];
  for (int i = tid; i < 128 * SL; i += 256) {
    int r = i / SL, s = i % SL;
    xt[i] = x[(size_t)(rowbase + r) * SL + s];
  }
  for (int i = tid; i < 128 * 64; i += 256) hbuf[i] = 0;
  __syncthreads();

  unsigned short* hw = hbuf + wave * 16 * 64;  // this wave's [16][64] h tile
  float c[4][8];
#pragma unroll
  for (int t = 0; t < 4; ++t)
#pragma unroll
    for (int r = 0; r < 8; ++r) c[t][r] = 0.f;

  for (int s = 0; s < SL; ++s) {
    const int st = dir ? (SL - 1 - s) : s;
    // A-fragments (ISA 16-bit A 16x32 layout): lane row = ln, K base = half*8
    const unsigned short* hr = hw + ln * 64 + half * 8;
    v8us l0 = *(const v8us*)(hr);
    v8us h0 = *(const v8us*)(hr + 16);
    v8us l1 = *(const v8us*)(hr + 32);
    v8us h1 = *(const v8us*)(hr + 48);
    v16bf a0 = as_bf16(__builtin_shufflevector(l0, h0, 0,1,2,3,4,5,6,7,8,9,10,11,12,13,14,15));
    v16bf a1 = as_bf16(__builtin_shufflevector(l1, h1, 0,1,2,3,4,5,6,7,8,9,10,11,12,13,14,15));
    // Tokens for the 8 C/D rows this lane holds (rows half*8 + r)
    int toks[8];
#pragma unroll
    for (int r = 0; r < 8; ++r)
      toks[r] = xt[(wave * 16 + half * 8 + r) * SL + st];

#pragma unroll
    for (int t = 0; t < 4; ++t) {           // 16 hidden cols per t
      v8f acc[4];                           // i, f, g, o gate tiles
#pragma unroll
      for (int g = 0; g < 4; ++g) {
        const int col = g * 64 + t * 16 + ln;
#pragma unroll
        for (int r = 0; r < 8; ++r) acc[g][r] = Tl[toks[r] * 256 + col]; // C = T[tok]
        const int nt = g * 4 + t;
        v16bf b0 = *(const v16bf*)(wfrag + ((0 * 16 + nt) * 32 + lane) * 16);
        acc[g] = __builtin_amdgcn_wmma_f32_16x16x32_bf16(false, a0, false, b0,
                                                         (short)0, acc[g], false, false);
        v16bf b1 = *(const v16bf*)(wfrag + ((1 * 16 + nt) * 32 + lane) * 16);
        acc[g] = __builtin_amdgcn_wmma_f32_16x16x32_bf16(false, a1, false, b1,
                                                         (short)0, acc[g], false, false);
      }
#pragma unroll
      for (int r = 0; r < 8; ++r) {
        float ig = sigmoid_(acc[0][r]);
        float fg = sigmoid_(acc[1][r]);
        float gg = tanh_(acc[2][r]);
        float og = sigmoid_(acc[3][r]);
        float cn = fg * c[t][r] + ig * gg;
        c[t][r] = cn;
        float hn = og * tanh_(cn);
        hw[(half * 8 + r) * 64 + t * 16 + ln] = f2bf(hn);
      }
    }
    __syncthreads();
  }

  // seq_feat: forward h -> cols 0..63, backward h -> cols 64..127
  for (int i = tid; i < 128 * 64; i += 256) {
    int r = i >> 6, hc = i & 63;
    feats[(size_t)(rowbase + r) * FD + dir * 64 + hc] = bf2f(hbuf[i]);
  }
}

// ---------------------------------------------------------------------------
// Kernel 2: per-sample features (entropy/length/char-dist/pattern + MLPs).
// Tiny work; one thread per sample, histogram in LDS.
// ---------------------------------------------------------------------------
__global__ __launch_bounds__(64) void feat_kernel(
    const int* __restrict__ x,
    const float* __restrict__ ew1, const float* __restrict__ eb1,
    const float* __restrict__ ew2, const float* __restrict__ eb2,
    const float* __restrict__ lw1, const float* __restrict__ lb1,
    const float* __restrict__ lw2, const float* __restrict__ lb2,
    const float* __restrict__ cw1, const float* __restrict__ cb1,
    const float* __restrict__ cw2, const float* __restrict__ cb2,
    const float* __restrict__ pw1, const float* __restrict__ pb1,
    const float* __restrict__ pw2, const float* __restrict__ pb2,
    float* __restrict__ feats) {
  __shared__ float cnt[64][NV];
  const int tid = threadIdx.x;
  const int b = blockIdx.x * 64 + tid;
  float* cn = cnt[tid];
#pragma unroll
  for (int v = 0; v < NV; ++v) cn[v] = 0.f;

  int len = 0, p1 = -1, p2 = -1, rep = 0, inc = 0, per = 0;
  for (int s = 0; s < SL; ++s) {
    int tok = x[(size_t)b * SL + s];
    if (tok != 0) {
      cn[tok] += 1.f;
      if (len >= 1) { rep += (tok == p1); inc += (tok > p1); }
      if (len >= 2) { per += (tok == p2); }
      p2 = p1; p1 = tok; ++len;
    }
  }
  const float lf = (float)len;
  const float d0 = fmaxf(lf, 1.f);
  float probs[NV];
  float ent = 0.f;
  int divc = 0;
#pragma unroll
  for (int v = 0; v < NV; ++v) {
    float p = cn[v] / d0;
    probs[v] = p;
    ent += p * __logf(p + 1e-8f);
    divc += (cn[v] > 0.f) ? 1 : 0;
  }
  ent = (len > 1) ? -ent : 0.f;

  float* fo = feats + (size_t)b * FD;
  { // entropy MLP 1->16->8
    float in = ent * 0.25f;
    float h[16];
#pragma unroll
    for (int j = 0; j < 16; ++j) h[j] = fmaxf(in * ew1[j] + eb1[j], 0.f);
#pragma unroll
    for (int o = 0; o < 8; ++o) {
      float s = eb2[o];
#pragma unroll
      for (int j = 0; j < 16; ++j) s += h[j] * ew2[j * 8 + o];
      fo[128 + o] = s;
    }
  }
  { // length MLP 1->16->8
    float in = lf / 60.f;
    float h[16];
#pragma unroll
    for (int j = 0; j < 16; ++j) h[j] = fmaxf(in * lw1[j] + lb1[j], 0.f);
#pragma unroll
    for (int o = 0; o < 8; ++o) {
      float s = lb2[o];
#pragma unroll
      for (int j = 0; j < 16; ++j) s += h[j] * lw2[j * 8 + o];
      fo[136 + o] = s;
    }
  }
  { // char-dist MLP 40->32->16
    float h[32];
#pragma unroll
    for (int j = 0; j < 32; ++j) {
      float s = cb1[j];
#pragma unroll
      for (int v = 0; v < NV; ++v) s += probs[v] * cw1[v * 32 + j];
      h[j] = fmaxf(s, 0.f);
    }
#pragma unroll
    for (int o = 0; o < 16; ++o) {
      float s = cb2[o];
#pragma unroll
      for (int j = 0; j < 32; ++j) s += h[j] * cw2[j * 16 + o];
      fo[144 + o] = s;
    }
  }
  { // pattern MLP 4->16->8
    float g = (len > 1) ? 1.f : 0.f;
    float d1 = fmaxf(lf - 1.f, 1.f), d2 = fmaxf(lf - 2.f, 1.f);
    float pin[4];
    pin[0] = g * (float)rep / d1;
    pin[1] = g * (float)inc / d1;
    pin[2] = g * (float)divc / d0;
    pin[3] = (len >= 4) ? g * (float)per / d2 : 0.f;
    float h[16];
#pragma unroll
    for (int j = 0; j < 16; ++j) {
      float s = pb1[j];
#pragma unroll
      for (int v = 0; v < 4; ++v) s += pin[v] * pw1[v * 16 + j];
      h[j] = fmaxf(s, 0.f);
    }
#pragma unroll
    for (int o = 0; o < 8; ++o) {
      float s = pb2[o];
#pragma unroll
      for (int j = 0; j < 16; ++j) s += h[j] * pw2[j * 8 + o];
      fo[160 + o] = s;
    }
  }
}

// ---------------------------------------------------------------------------
// Kernel 3: generic WMMA GEMM  C = act(A[M,K] @ W[K,N] + bias), M = NB.
// W staged in LDS as zero-padded bf16 fragments; bias used as C-init.
// Templated on tile counts so accumulator indexing is compile-time.
// ---------------------------------------------------------------------------
template <int KC, int NT, bool RELU>
__global__ __launch_bounds__(256) void gemm_bias_act_kernel(
    const float* __restrict__ A, const float* __restrict__ W,
    const float* __restrict__ bias, float* __restrict__ C, int K, int N) {
  extern __shared__ char smem[];
  unsigned short* wfrag = (unsigned short*)smem;           // KC*NT*512 bf16
  float* bl = (float*)(smem + (size_t)KC * NT * 512 * 2);  // NT*16 bias (padded)
  const int tid = threadIdx.x;
  const int wave = tid >> 5, lane = tid & 31, half = lane >> 4, ln = lane & 15;
  const int rowbase = blockIdx.x * 128;

  for (int i = tid; i < KC * NT * 512; i += 256) {
    int e = i & 15, l = (i >> 4) & 31, f = i >> 9;
    int nt = f % NT, kc = f / NT;
    int k = kc * 32 + (l >> 4) * 16 + e;
    int n = nt * 16 + (l & 15);
    wfrag[i] = (k < K && n < N) ? f2bf(W[(size_t)k * N + n]) : (unsigned short)0;
  }
  for (int i = tid; i < NT * 16; i += 256) bl[i] = (i < N) ? bias[i] : 0.f;
  __syncthreads();

  const int arow = rowbase + wave * 16 + ln;  // A-matrix row this lane feeds
  v8f acc[NT];
#pragma unroll
  for (int nt = 0; nt < NT; ++nt) {
    float bv = bl[nt * 16 + ln];
#pragma unroll
    for (int r = 0; r < 8; ++r) acc[nt][r] = bv;
  }
#pragma unroll
  for (int kc = 0; kc < KC; ++kc) {
    v16us au;
#pragma unroll
    for (int e = 0; e < 16; ++e) {
      int k = kc * 32 + half * 16 + e;
      float v = (k < K) ? A[(size_t)arow * K + k] : 0.f;
      au[e] = f2bf(v);
    }
    v16bf a = as_bf16(au);
#pragma unroll
    for (int nt = 0; nt < NT; ++nt) {
      v16bf b = *(const v16bf*)(wfrag + (((size_t)kc * NT + nt) * 32 + lane) * 16);
      acc[nt] = __builtin_amdgcn_wmma_f32_16x16x32_bf16(false, a, false, b,
                                                        (short)0, acc[nt], false, false);
    }
  }
#pragma unroll
  for (int nt = 0; nt < NT; ++nt) {
#pragma unroll
    for (int r = 0; r < 8; ++r) {
      int n = nt * 16 + ln;
      if (n < N) {
        float v = acc[nt][r];
        if (RELU) v = fmaxf(v, 0.f);
        C[(size_t)(rowbase + wave * 16 + half * 8 + r) * N + n] = v;
      }
    }
  }
}

// ---------------------------------------------------------------------------
// Kernel 4: row softmax over E=4 logits.
// ---------------------------------------------------------------------------
__global__ void softmax_kernel(const float* __restrict__ logits,
                               float* __restrict__ out) {
  int b = blockIdx.x * 256 + threadIdx.x;
  if (b >= NB) return;
  float l0 = logits[b * 4 + 0], l1 = logits[b * 4 + 1];
  float l2 = logits[b * 4 + 2], l3 = logits[b * 4 + 3];
  float m = fmaxf(fmaxf(l0, l1), fmaxf(l2, l3));
  float e0 = __expf(l0 - m), e1 = __expf(l1 - m);
  float e2 = __expf(l2 - m), e3 = __expf(l3 - m);
  float inv = 1.f / (e0 + e1 + e2 + e3);
  out[b * 4 + 0] = e0 * inv;
  out[b * 4 + 1] = e1 * inv;
  out[b * 4 + 2] = e2 * inv;
  out[b * 4 + 3] = e3 * inv;
}

// ---------------------------------------------------------------------------
template <int KC, int NT, bool RELU>
static void launch_gemm(const float* A, const float* W, const float* bias,
                        float* C, int K, int N, hipStream_t s) {
  size_t sh = (size_t)KC * NT * 512 * 2 + (size_t)NT * 16 * 4;
  gemm_bias_act_kernel<KC, NT, RELU><<<NB / 128, 256, sh, s>>>(A, W, bias, C, K, N);
}

extern "C" void kernel_launch(void* const* d_in, const int* in_sizes, int n_in,
                              void* d_out, int out_size, void* d_ws, size_t ws_size,
                              hipStream_t stream) {
  (void)in_sizes; (void)n_in; (void)out_size; (void)ws_size;
  const int*   x     = (const int*)d_in[0];
  const float* emb   = (const float*)d_in[1];
  const float* wih_f = (const float*)d_in[2];
  const float* whh_f = (const float*)d_in[3];
  const float* bih_f = (const float*)d_in[4];
  const float* bhh_f = (const float*)d_in[5];
  const float* wih_b = (const float*)d_in[6];
  const float* whh_b = (const float*)d_in[7];
  const float* bih_b = (const float*)d_in[8];
  const float* bhh_b = (const float*)d_in[9];
  const float* ew1 = (const float*)d_in[10]; const float* eb1 = (const float*)d_in[11];
  const float* ew2 = (const float*)d_in[12]; const float* eb2 = (const float*)d_in[13];
  const float* lw1 = (const float*)d_in[14]; const float* lb1 = (const float*)d_in[15];
  const float* lw2 = (const float*)d_in[16]; const float* lb2 = (const float*)d_in[17];
  const float* cw1 = (const float*)d_in[18]; const float* cb1 = (const float*)d_in[19];
  const float* cw2 = (const float*)d_in[20]; const float* cb2 = (const float*)d_in[21];
  const float* pw1 = (const float*)d_in[22]; const float* pb1 = (const float*)d_in[23];
  const float* pw2 = (const float*)d_in[24]; const float* pb2 = (const float*)d_in[25];
  const float* projw = (const float*)d_in[26]; const float* projb = (const float*)d_in[27];
  const float* wv = (const float*)d_in[28]; const float* bv = (const float*)d_in[29];
  const float* wo = (const float*)d_in[30]; const float* bo = (const float*)d_in[31];
  const float* g1w = (const float*)d_in[32]; const float* g1b = (const float*)d_in[33];
  const float* g2w = (const float*)d_in[34]; const float* g2b = (const float*)d_in[35];
  const float* g3w = (const float*)d_in[36]; const float* g3b = (const float*)d_in[37];
  const float* g4w = (const float*)d_in[38]; const float* g4b = (const float*)d_in[39];

  float* ws = (float*)d_ws;
  float* Tf    = ws;                       // 40*256
  float* Tb    = Tf + 40 * 256;            // 40*256
  float* feats = Tb + 40 * 256;            // NB*FD
  float* bufA  = feats + (size_t)NB * FD;  // NB*FD
  float* bufB  = bufA + (size_t)NB * FD;   // NB*FD

  // Token->gate-preactivation tables (folds embedding + input GEMM + biases)
  build_table_kernel<<<NV, 256, 0, stream>>>(emb, wih_f, bih_f, bhh_f, Tf);
  build_table_kernel<<<NV, 256, 0, stream>>>(emb, wih_b, bih_b, bhh_b, Tb);

  // Fused bidirectional LSTM (WMMA bf16 recurrence)
  lstm_kernel<<<dim3(NB / 128, 2), 256, LSTM_LDS, stream>>>(x, whh_f, whh_b, Tf, Tb, feats);

  // Scalar per-sample features into feats[:,128:168]
  feat_kernel<<<NB / 64, 64, 0, stream>>>(x, ew1, eb1, ew2, eb2, lw1, lb1, lw2, lb2,
                                          cw1, cb1, cw2, cb2, pw1, pb1, pw2, pb2, feats);

  // Head: proj -> v -> o -> 128 -> 64 -> 32 -> 4 (WMMA GEMMs, ping-pong buffers)
  launch_gemm<6, 11, false>(feats, projw, projb, bufA, FD, FD, stream);
  launch_gemm<6, 11, false>(bufA, wv, bv, bufB, FD, FD, stream);
  launch_gemm<6, 11, false>(bufB, wo, bo, bufA, FD, FD, stream);
  launch_gemm<6, 8,  true >(bufA, g1w, g1b, bufB, FD, 128, stream);
  launch_gemm<4, 4,  true >(bufB, g2w, g2b, bufA, 128, 64, stream);
  launch_gemm<2, 2,  true >(bufA, g3w, g3b, bufB, 64, 32, stream);
  launch_gemm<1, 1,  false>(bufB, g4w, g4b, bufA, 32, NE, stream);

  softmax_kernel<<<NB / 256, 256, 0, stream>>>(bufA, (float*)d_out);
}
